// Encoder_73778948211333
// MI455X (gfx1250) — matrix operand order repaired and checked by reading, not compiled, not run on hardware
//
#include <hip/hip_runtime.h>

typedef __attribute__((ext_vector_type(16))) _Float16 v16h;
typedef __attribute__((ext_vector_type(8)))  _Float16 v8h;
typedef __attribute__((ext_vector_type(4)))  _Float16 h4;
typedef __attribute__((ext_vector_type(8)))  float    v8f;

#define D_MODEL 512
#define N_HEADS 8
#define DKH     64
#define D_FF    2048
#define SEQ     2048
#define BATCH   4
#define ROWS    (BATCH * SEQ)   // 8192
#define STACK   6

// --- gfx1250 async copy to LDS (async/tensor engine path), with fallback ----
#if __has_builtin(__builtin_amdgcn_global_load_async_to_lds_b128)
#define HAVE_ASYNC 1
typedef int v4i_async __attribute__((__vector_size__(4 * sizeof(int))));
#define ASYNC_CP16(gp, lp)                                          \
  __builtin_amdgcn_global_load_async_to_lds_b128((v4i_async*)(gp),  \
                                                 (v4i_async*)(lp), 0, 0)
#define WAIT_ASYNC(n) asm volatile("s_wait_asynccnt %0" ::"n"(n) : "memory")
#else
#define HAVE_ASYNC 0
#define ASYNC_CP16(gp, lp) (*(float4*)(lp) = *(const float4*)(gp))
#define WAIT_ASYNC(n) ((void)0)
#endif

static __device__ __forceinline__ v16h cat8(v8h lo, v8h hi) {
  return __builtin_shufflevector(lo, hi, 0, 1, 2, 3, 4, 5, 6, 7,
                                 8, 9, 10, 11, 12, 13, 14, 15);
}

// butterfly xor-lane exchange via ds_swizzle_b32 (group-of-32 mode:
// offset = xor_mask<<10 | or_mask<<5 | and_mask; and_mask=0x1f keeps all lanes)
template <int M>
static __device__ __forceinline__ float xswz(float v) {
  return __int_as_float(
      __builtin_amdgcn_ds_swizzle(__float_as_int(v), (M << 10) | 0x1f));
}
// max-reduce across 16-lane half groups (xor masks < 16 stay in-half)
static __device__ __forceinline__ float redmax16(float v) {
  v = fmaxf(v, xswz<8>(v)); v = fmaxf(v, xswz<4>(v));
  v = fmaxf(v, xswz<2>(v)); v = fmaxf(v, xswz<1>(v));
  return v;
}
static __device__ __forceinline__ float redsum16(float v) {
  v += xswz<8>(v); v += xswz<4>(v); v += xswz<2>(v); v += xswz<1>(v);
  return v;
}
static __device__ __forceinline__ float redsum32(float v) {
  v += xswz<16>(v); v += xswz<8>(v); v += xswz<4>(v);
  v += xswz<2>(v);  v += xswz<1>(v);
  return v;
}

// ---------------------------------------------------------------------------
// WMMA 16x16x32 f16 fragment layouts (wave32):
//  A (16x32): lane L -> row m=L&15; frag elems 0..7  = k (L>>4)*8 + 0..7,
//             elems 8..15 = k 16 + (L>>4)*8 + 0..7   (two contiguous runs)
//  B (32x16): lane L -> col n=L&15; frag elems 0..15 = same k order as A (P@V uses k=(L>>4)*16+e for A and B)
//  C (16x16): lane L -> col n=L&15; VGPR r -> row m=(L>>4)*8 + r
// ---------------------------------------------------------------------------

// ===========================================================================
// Bulk f32 -> f16 convert (weights once per call; initial x -> xh).
// ===========================================================================
__launch_bounds__(256)
__global__ void cvt_kernel(const float* __restrict__ s, _Float16* __restrict__ d,
                           long n) {
  const long i = ((long)blockIdx.x * 256 + threadIdx.x) * 8;
  if (i >= n) return;
  float4 a = *(const float4*)(s + i);
  float4 b = *(const float4*)(s + i + 4);
  v8h o;
  o[0] = (_Float16)a.x; o[1] = (_Float16)a.y;
  o[2] = (_Float16)a.z; o[3] = (_Float16)a.w;
  o[4] = (_Float16)b.x; o[5] = (_Float16)b.y;
  o[6] = (_Float16)b.z; o[7] = (_Float16)b.w;
  *(v8h*)(d + i) = o;
}

// ===========================================================================
// Flash attention (q=k=v = head slice of xh, all f16).  One wave per
// (b,h,16-query tile); 32-key chunks, double-buffered async LDS staging.
// ===========================================================================
#define KVS 72   // padded LDS row stride in f16 (144B = 9*16B)

__launch_bounds__(32)
__global__ void attn_kernel(const _Float16* __restrict__ xh,
                            _Float16* __restrict__ y) {
  const int qt   = blockIdx.x;          // 0..127
  const int h    = blockIdx.y;          // 0..7
  const int b    = blockIdx.z;          // 0..3
  const int lane = threadIdx.x;
  const int half = lane >> 4;
  const int l16  = lane & 15;
  const int q0   = qt * 16;

  __shared__ _Float16 Ks[2][32 * KVS];  // K/V tile f16 [key][dk]
  __shared__ _Float16 pbuf[16 * 40];    // P staging (C->A layout)

  // --- Q fragments (A layout, b128 global loads), scaled by 1/8 ---
  v16h aq0, aq1;
  {
    const _Float16* qrow =
        xh + ((size_t)(b * SEQ + q0 + l16) * D_MODEL) + h * DKH + half * 8;
    aq0 = cat8(*(const v8h*)(qrow), *(const v8h*)(qrow + 16));
    aq1 = cat8(*(const v8h*)(qrow + 32), *(const v8h*)(qrow + 48));
#pragma unroll
    for (int e = 0; e < 16; ++e) {
      aq0[e] = aq0[e] * (_Float16)0.125f;   // exact power-of-two scale
      aq1[e] = aq1[e] * (_Float16)0.125f;
    }
  }

  v8f oacc[4] = {v8f{}, v8f{}, v8f{}, v8f{}};
  float mrow[8], lrow[8];
#pragma unroll
  for (int r = 0; r < 8; ++r) { mrow[r] = -1e30f; lrow[r] = 0.0f; }

  // async stage one 32x64 f16 K/V tile (8 x b128 per lane)
  auto issue_kv = [&](int kt, int buf) {
#pragma unroll
    for (int i = 0; i < 8; ++i) {
      const int e   = i * 32 + lane;
      const int row = e >> 3;            // 8 b128 per 64-f16 row
      const int c8  = (e & 7) * 8;
      const _Float16* g =
          xh + (size_t)(b * SEQ + kt + row) * D_MODEL + h * DKH + c8;
      ASYNC_CP16(g, &Ks[buf][row * KVS + c8]);
    }
  };

  issue_kv(0, 0);
  int cur = 0;
  for (int kt = 0; kt < SEQ; kt += 32) {
    if (kt + 32 < SEQ) { issue_kv(kt + 32, cur ^ 1); WAIT_ASYNC(8); }
    else               { WAIT_ASYNC(0); }
    __syncthreads();
    const _Float16* kv = Ks[cur];

    // --- scores S = (Q/8) @ K^T over dk=64 (2 WMMA per 16-key half) ---
    v8f s[2];
#pragma unroll
    for (int t = 0; t < 2; ++t) {
      const _Float16* krow = kv + (t * 16 + l16) * KVS + half * 8;
      v16h bk0 = cat8(*(const v8h*)(krow), *(const v8h*)(krow + 16));
      v16h bk1 = cat8(*(const v8h*)(krow + 32), *(const v8h*)(krow + 48));
      v8f c = {};
      c = __builtin_amdgcn_wmma_f32_16x16x32_f16(false, aq0, false, bk0,
                                                 (short)0, c, false, false);
      c = __builtin_amdgcn_wmma_f32_16x16x32_f16(false, aq1, false, bk1,
                                                 (short)0, c, false, false);
      s[t] = c;
    }

    // --- online softmax over the 32-key chunk ---
#pragma unroll
    for (int r = 0; r < 8; ++r) {
      float v0 = s[0][r], v1 = s[1][r];
      const float mx   = redmax16(fmaxf(v0, v1));
      const float mnew = fmaxf(mrow[r], mx);
      const float alpha = __expf(mrow[r] - mnew);
      const float p0 = __expf(v0 - mnew);
      const float p1 = __expf(v1 - mnew);
      const float rs = redsum16(p0 + p1);
      lrow[r] = lrow[r] * alpha + rs;
      mrow[r] = mnew;
      oacc[0][r] *= alpha; oacc[1][r] *= alpha;
      oacc[2][r] *= alpha; oacc[3][r] *= alpha;
      const int m = half * 8 + r;
      pbuf[m * 40 + l16]      = (_Float16)p0;
      pbuf[m * 40 + 16 + l16] = (_Float16)p1;
    }
    __syncthreads();

    // --- gather P as A fragment (two b128 LDS loads) ---
    const _Float16* pr = &pbuf[l16 * 40 + half * 16];
    v16h pa = cat8(*(const v8h*)(pr), *(const v8h*)(pr + 8));

    // --- O += P @ V (keys gathered column-wise, d16/d16_hi loads) ---
#pragma unroll
    for (int f = 0; f < 4; ++f) {
      const int n = f * 16 + l16;
      v16h bv;
#pragma unroll
      for (int j = 0; j < 8; ++j) {
        const int kj = half * 16 + 2 * j;
        bv[2 * j]     = kv[kj * KVS + n];
        bv[2 * j + 1] = kv[(kj + 1) * KVS + n];
      }
      oacc[f] = __builtin_amdgcn_wmma_f32_16x16x32_f16(false, pa, false, bv,
                                                       (short)0, oacc[f],
                                                       false, false);
    }
    cur ^= 1;
  }

  // --- normalize and write out f16 (heads concatenated) ---
#pragma unroll
  for (int r = 0; r < 8; ++r) {
    const float inv = 1.0f / lrow[r];
    const int m = half * 8 + r;
    _Float16* yr = y + ((size_t)(b * SEQ + q0 + m) * D_MODEL) + h * DKH + l16;
    yr[0]  = (_Float16)(oacc[0][r] * inv);
    yr[16] = (_Float16)(oacc[1][r] * inv);
    yr[32] = (_Float16)(oacc[2][r] * inv);
    yr[48] = (_Float16)(oacc[3][r] * inv);
  }
}

// ===========================================================================
// Tiled f16 GEMM: acc = X @ W^T (f32 accum); epilogue adds bias (+residual),
// optional f32 and/or f16 outputs.  128 threads (4 waves); 128x64 block tile;
// K-step 64 (16 WMMA/wave/step); double-buffered async f16 LDS tiles.
// ===========================================================================
#define GS 72   // padded LDS row stride in f16 (144B = 9*16B)

__launch_bounds__(128)
__global__ void gemm_kernel(const _Float16* __restrict__ X,
                            const _Float16* __restrict__ W,
                            const float* __restrict__ bias,
                            const float* __restrict__ res,
                            float* __restrict__ outF,
                            _Float16* __restrict__ outH, int K, int N) {
  const int n0   = blockIdx.x * 64;
  const int r0   = blockIdx.y * 128;
  const int tid  = threadIdx.x;
  const int lane = tid & 31;
  const int wave = tid >> 5;
  const int half = lane >> 4;
  const int l16  = lane & 15;

  __shared__ _Float16 Xs[2][128 * GS];   // 2 x 18KB
  __shared__ _Float16 Ws[2][64 * GS];    // 2 x 9KB

  v8f acc[2][4] = {{v8f{}, v8f{}, v8f{}, v8f{}}, {v8f{}, v8f{}, v8f{}, v8f{}}};

  // async stage one 64-wide K-step: X 128x64 f16 (8 b128/thr), W 64x64 (4/thr)
  auto issue_tile = [&](int kk, int buf) {
#pragma unroll
    for (int i = 0; i < 8; ++i) {
      const int e   = i * 128 + tid;
      const int row = e >> 3;            // 8 b128 per 64-f16 row
      const int c8  = (e & 7) * 8;
      ASYNC_CP16(X + (size_t)(r0 + row) * K + kk + c8, &Xs[buf][row * GS + c8]);
    }
#pragma unroll
    for (int i = 0; i < 4; ++i) {
      const int e   = i * 128 + tid;
      const int row = e >> 3;
      const int c8  = (e & 7) * 8;
      ASYNC_CP16(W + (size_t)(n0 + row) * K + kk + c8, &Ws[buf][row * GS + c8]);
    }
  };

  issue_tile(0, 0);
  int cur = 0;
  for (int kk = 0; kk < K; kk += 64) {
    __syncthreads();   // all waves done reading the buffer being re-filled
    if (kk + 64 < K) { issue_tile(kk + 64, cur ^ 1); WAIT_ASYNC(12); }
    else             { WAIT_ASYNC(0); }
    __syncthreads();   // this K-step's tile visible to all waves

#pragma unroll
    for (int kh = 0; kh < 2; ++kh) {
      // two A fragments (wave's 32 rows), two b128 LDS loads each
      v16h a[2];
#pragma unroll
      for (int sub = 0; sub < 2; ++sub) {
        const _Float16* ap =
            &Xs[cur][(wave * 32 + sub * 16 + l16) * GS + kh * 32 + half * 8];
        a[sub] = cat8(*(const v8h*)(ap), *(const v8h*)(ap + 16));
      }
      // 4 B fragments, each reused by both A fragments
#pragma unroll
      for (int f = 0; f < 4; ++f) {
        const _Float16* bp =
            &Ws[cur][(f * 16 + l16) * GS + kh * 32 + half * 8];
        v16h bf = cat8(*(const v8h*)(bp), *(const v8h*)(bp + 16));
        acc[0][f] = __builtin_amdgcn_wmma_f32_16x16x32_f16(
            false, a[0], false, bf, (short)0, acc[0][f], false, false);
        acc[1][f] = __builtin_amdgcn_wmma_f32_16x16x32_f16(
            false, a[1], false, bf, (short)0, acc[1][f], false, false);
      }
    }
    cur ^= 1;
  }

  // epilogue: bias, optional residual, optional f32 / f16 stores
#pragma unroll
  for (int sub = 0; sub < 2; ++sub) {
#pragma unroll
    for (int f = 0; f < 4; ++f) {
      const int n = n0 + f * 16 + l16;
      const float bv = bias[n];
#pragma unroll
      for (int r = 0; r < 8; ++r) {
        const int row = r0 + wave * 32 + sub * 16 + half * 8 + r;
        const size_t idx = (size_t)row * N + n;
        float v = acc[sub][f][r] + bv;
        if (res)  v += res[idx];
        if (outF) outF[idx] = v;
        if (outH) outH[idx] = (_Float16)v;
      }
    }
  }
}

// ===========================================================================
// LayerNorm over D_MODEL=512, one wave per row; updates x (f32, in place)
// and emits the f16 copy xh for downstream WMMA consumers.
// ===========================================================================
__launch_bounds__(32)
__global__ void ln_kernel(float* __restrict__ x, const float* __restrict__ g,
                          const float* __restrict__ bta,
                          _Float16* __restrict__ xh) {
  const int row  = blockIdx.x;
  const int lane = threadIdx.x;
  float* xr = x + (size_t)row * D_MODEL;

  float4 v[4];
  float s = 0.0f, ss = 0.0f;
#pragma unroll
  for (int c = 0; c < 4; ++c) {
    v[c] = *(const float4*)(xr + c * 128 + lane * 4);
    s  += v[c].x + v[c].y + v[c].z + v[c].w;
    ss += v[c].x * v[c].x + v[c].y * v[c].y + v[c].z * v[c].z + v[c].w * v[c].w;
  }
  s  = redsum32(s);
  ss = redsum32(ss);
  const float mu  = s * (1.0f / 512.0f);
  const float var = ss * (1.0f / 512.0f) - mu * mu;
  const float rs  = rsqrtf(var + 1e-8f);
#pragma unroll
  for (int c = 0; c < 4; ++c) {
    const float4 gg = *(const float4*)(g + c * 128 + lane * 4);
    const float4 bb = *(const float4*)(bta + c * 128 + lane * 4);
    float4 o;
    o.x = (v[c].x - mu) * rs * gg.x + bb.x;
    o.y = (v[c].y - mu) * rs * gg.y + bb.y;
    o.z = (v[c].z - mu) * rs * gg.z + bb.z;
    o.w = (v[c].w - mu) * rs * gg.w + bb.w;
    *(float4*)(xr + c * 128 + lane * 4) = o;
    h4 oh;
    oh[0] = (_Float16)o.x; oh[1] = (_Float16)o.y;
    oh[2] = (_Float16)o.z; oh[3] = (_Float16)o.w;
    *(h4*)(xh + (size_t)row * D_MODEL + c * 128 + lane * 4) = oh;
  }
}

// ===========================================================================
extern "C" void kernel_launch(void* const* d_in, const int* in_sizes, int n_in,
                              void* d_out, int out_size, void* d_ws, size_t ws_size,
                              hipStream_t stream) {
  (void)in_sizes; (void)n_in; (void)out_size; (void)ws_size;
  const float* x_in  = (const float*)d_in[0];
  const float* wo_w  = (const float*)d_in[1];
  const float* wo_b  = (const float*)d_in[2];
  const float* ln1_g = (const float*)d_in[3];
  const float* ln1_b = (const float*)d_in[4];
  const float* ln2_g = (const float*)d_in[5];
  const float* ln2_b = (const float*)d_in[6];
  const float* fc1_w = (const float*)d_in[7];
  const float* fc1_b = (const float*)d_in[8];
  const float* fc2_w = (const float*)d_in[9];
  const float* fc2_b = (const float*)d_in[10];

  float* x = (float*)d_out;                       // f32 residual stream

  // f16 workspace layout
  _Float16* xh    = (_Float16*)d_ws;              // 8192x512
  _Float16* y16   = xh  + (size_t)ROWS * D_MODEL; // 8192x512
  _Float16* h16   = y16 + (size_t)ROWS * D_MODEL; // 8192x2048
  _Float16* w_wo  = h16 + (size_t)ROWS * D_FF;    // 6x512x512
  _Float16* w_fc1 = w_wo  + (size_t)STACK * D_MODEL * D_MODEL;  // 6x2048x512
  _Float16* w_fc2 = w_fc1 + (size_t)STACK * D_FF * D_MODEL;     // 6x512x2048

  auto cvt = [&](const float* s, _Float16* d, long n) {
    cvt_kernel<<<(unsigned)((n / 8 + 255) / 256), 256, 0, stream>>>(s, d, n);
  };

  // all weights -> f16 once per call (deterministic)
  cvt(wo_w,  w_wo,  (long)STACK * D_MODEL * D_MODEL);
  cvt(fc1_w, w_fc1, (long)STACK * D_FF * D_MODEL);
  cvt(fc2_w, w_fc2, (long)STACK * D_MODEL * D_FF);

  (void)hipMemcpyAsync(x, x_in, sizeof(float) * (size_t)ROWS * D_MODEL,
                       hipMemcpyDeviceToDevice, stream);
  cvt(x, xh, (long)ROWS * D_MODEL);

  const dim3 attnGrid(SEQ / 16, N_HEADS, BATCH);
  const dim3 lnGrid(ROWS);
  for (int i = 0; i < STACK; ++i) {
    attn_kernel<<<attnGrid, 32, 0, stream>>>(xh, y16);
    // x = y @ wo^T + wo_b + x   (residual fused, f32 out)
    gemm_kernel<<<dim3(D_MODEL / 64, ROWS / 128), 128, 0, stream>>>(
        y16, w_wo + (size_t)i * D_MODEL * D_MODEL, wo_b + i * D_MODEL, x, x,
        nullptr, D_MODEL, D_MODEL);
    ln_kernel<<<lnGrid, 32, 0, stream>>>(x, ln1_g + i * D_MODEL,
                                         ln1_b + i * D_MODEL, xh);
    // h16 = x @ fc1^T + fc1_b   (f16 out only; no activation per reference)
    gemm_kernel<<<dim3(D_FF / 64, ROWS / 128), 128, 0, stream>>>(
        xh, w_fc1 + (size_t)i * D_FF * D_MODEL, fc1_b + i * D_FF, nullptr,
        nullptr, h16, D_MODEL, D_FF);
    // x = h @ fc2^T + fc2_b + x   (residual fused, f32 out)
    gemm_kernel<<<dim3(D_MODEL / 64, ROWS / 128), 128, 0, stream>>>(
        h16, w_fc2 + (size_t)i * D_MODEL * D_FF, fc2_b + i * D_MODEL, x, x,
        nullptr, D_FF, D_MODEL);
    ln_kernel<<<lnGrid, 32, 0, stream>>>(x, ln2_g + i * D_MODEL,
                                         ln2_b + i * D_MODEL, xh);
  }
}
